// proteusAI_19825569038955
// MI455X (gfx1250) — matrix operand order repaired and checked by reading, not compiled, not run on hardware
//
#include <hip/hip_runtime.h>

// ---------------------------------------------------------------------------
// geo-attention for MI455X (gfx1250, wave32, WMMA f16->f32)
// B=2, N=2048, D=512, H=8, DK=64
// ---------------------------------------------------------------------------

#define B_  2
#define N_  2048
#define D_  512
#define H_  8
#define DK_ 64

typedef __attribute__((ext_vector_type(16))) _Float16 v16h;
typedef __attribute__((ext_vector_type(8)))  _Float16 v8h;
typedef __attribute__((ext_vector_type(8)))  float    v8f;

__device__ __forceinline__ v8f wmma16(v16h a, v16h b, v8f c) {
  return __builtin_amdgcn_wmma_f32_16x16x32_f16(false, a, false, b, (short)0, c,
                                                false, false);
}

// A-fragment (16-bit, 16x32): lanes 0-15 rows M=0..15 with K = 0..15,
// lanes 16-31 same rows with K = 16..31.  p points at K=0 of the row.
__device__ __forceinline__ v16h load_a(const _Float16* p, int hw) {
  v8h lo = *(const v8h*)(p + hw * 16);
  v8h hi = *(const v8h*)(p + hw * 16 + 8);
  v16h r;
#pragma unroll
  for (int i = 0; i < 8; ++i) { r[i] = lo[i]; r[i + 8] = hi[i]; }
  return r;
}

__device__ __forceinline__ void wave_lds_fence() {
  asm volatile("s_wait_dscnt 0" ::: "memory");
}

// ---------------------------------------------------------------------------
// Kernel 0: weight conversion.  projT[t][h][dk][d] = (f16) proj_t[h][d][dk]
//           w16[i][j]          = (f16) out_w[i][j]
// ---------------------------------------------------------------------------
__global__ void convert_kernel(const float* __restrict__ qp,
                               const float* __restrict__ kp,
                               const float* __restrict__ vp,
                               const float* __restrict__ ow,
                               _Float16* __restrict__ projT,
                               _Float16* __restrict__ w16) {
  int idx = blockIdx.x * blockDim.x + threadIdx.x;
  const int total_p = 3 * H_ * DK_ * D_;  // 786432 = 3 * 2^18
  if (idx < total_p) {
    int d  = idx & (D_ - 1);
    int dk = (idx >> 9) & (DK_ - 1);
    int h  = (idx >> 15) & (H_ - 1);
    int t  = idx >> 18;
    const float* src = (t == 0) ? qp : (t == 1) ? kp : vp;
    projT[idx] = (_Float16)src[((size_t)h * D_ + d) * DK_ + dk];
  }
  int widx = idx - total_p;
  if (widx >= 0 && widx < D_ * D_) w16[widx] = (_Float16)ow[widx];
}

// ---------------------------------------------------------------------------
// Kernel 1: per-head projections.  One wave computes a 16xDK (16x64) tile.
//   t = blockIdx.y selects q/k/v.
//   Q,K stored row-major f16 [b][h][n][64];  V stored transposed [b][h][64][n].
// ---------------------------------------------------------------------------
__global__ __launch_bounds__(128) void proj_kernel(
    const float* __restrict__ qx, const float* __restrict__ kx,
    const float* __restrict__ vx, const float* __restrict__ qb,
    const float* __restrict__ kb, const float* __restrict__ vb,
    const _Float16* __restrict__ projT, _Float16* __restrict__ Qh,
    _Float16* __restrict__ Kh, _Float16* __restrict__ VT) {
  const int lane = threadIdx.x & 31;
  const int wv   = threadIdx.x >> 5;
  const int hw   = lane >> 4;
  const int l16  = lane & 15;
  const int t    = blockIdx.y;

  int tile = blockIdx.x * 4 + wv;          // 0 .. 2047
  int qt   = tile & 127;
  int h    = (tile >> 7) & (H_ - 1);
  int b    = tile >> 10;
  int row0 = qt * 16;

  const float* x    = (t == 0) ? qx : (t == 1) ? kx : vx;
  const float* bias = (t == 0) ? qb : (t == 1) ? kb : vb;
  const _Float16* pT = projT + ((size_t)t * H_ + h) * DK_ * D_;

  v8f acc[4] = {};
  for (int c = 0; c < D_ / 32; ++c) {       // 16 K-chunks of 32
    // issue all B-fragment loads as one clause, then the A gather, then WMMAs
    v16h bf[4];
#pragma unroll
    for (int j = 0; j < 4; ++j)
      bf[j] = *(const v16h*)(pT + (size_t)(j * 16 + l16) * D_ + c * 32 + hw * 16);

    const float* xr = x + ((size_t)b * N_ + row0 + l16) * D_ + c * 32;
    v16h a;
#pragma unroll
    for (int i = 0; i < 8; ++i) {
      a[i]     = (_Float16)xr[hw * 16 + i];
      a[i + 8] = (_Float16)xr[hw * 16 + 8 + i];
    }
#pragma unroll
    for (int j = 0; j < 4; ++j) acc[j] = wmma16(a, bf[j], acc[j]);
  }

  size_t hb = ((size_t)b * H_ + h);
#pragma unroll
  for (int j = 0; j < 4; ++j) {
    float bv = bias[h * DK_ + j * 16 + l16];
#pragma unroll
    for (int r = 0; r < 8; ++r) {
      float val = acc[j][r] + bv;
      int row = row0 + r + 8 * hw;
      int dk  = j * 16 + l16;
      if (t == 2) {
        VT[(hb * DK_ + dk) * N_ + row] = (_Float16)val;
      } else {
        _Float16* dst = (t == 0) ? Qh : Kh;
        dst[(hb * N_ + row) * DK_ + dk] = (_Float16)val;
      }
    }
  }
}

// ---------------------------------------------------------------------------
// Kernel 2: flash attention with geometric RBF bias + validity window.
// One wave owns a 16-query tile, streams keys in tiles of 32.
// ---------------------------------------------------------------------------
__global__ __launch_bounds__(128) void attn_kernel(
    const float* __restrict__ coords, const float* __restrict__ spreads,
    const unsigned char* __restrict__ maskp, const _Float16* __restrict__ Qh,
    const _Float16* __restrict__ Kh, const _Float16* __restrict__ VT,
    _Float16* __restrict__ inter) {
  __shared__ __align__(32) _Float16 ldsP[4][16][32];

  const int lane = threadIdx.x & 31;
  const int wv   = threadIdx.x >> 5;
  const int hw   = lane >> 4;
  const int l16  = lane & 15;

  int tile = blockIdx.x * 4 + wv;          // 0 .. 2047
  int qt   = tile & 127;
  int h    = (tile >> 7) & (H_ - 1);
  int b    = tile >> 10;
  int q0   = qt * 16;

  float spread = spreads[h];
  float inv2s2 = 1.0f / (2.0f * spread * spread);
  float d2min  = 0.010050336f / inv2s2;     // -ln(0.99)/inv2s2  (rbf <= 0.99)
  float d2max  = 4.6051702f / inv2s2;       // -ln(0.01)/inv2s2  (rbf >= 0.01)
  float betaC  = -2.0f * inv2s2;            // BETA * (-inv2s2)

  size_t hb = ((size_t)b * H_ + h);
  const _Float16* Qb    = Qh + (hb * N_ + q0) * DK_;
  const _Float16* Kbase = Kh + hb * N_ * DK_;
  const _Float16* Vbase = VT + hb * DK_ * (size_t)N_;
  const float* cbase    = coords + (size_t)b * N_ * 3;
  const unsigned char* mbase = maskp + (size_t)b * N_;

  v16h aq0 = load_a(Qb + (size_t)l16 * DK_, hw);        // K = 0..31
  v16h aq1 = load_a(Qb + (size_t)l16 * DK_ + 32, hw);   // K = 32..63

  float qcx[8], qcy[8], qcz[8];
#pragma unroll
  for (int r = 0; r < 8; ++r) {
    const float* c = cbase + (size_t)(q0 + r + 8 * hw) * 3;
    qcx[r] = c[0]; qcy[r] = c[1]; qcz[r] = c[2];
  }

  float m[8], l[8];
  v8f O[4] = {};
#pragma unroll
  for (int r = 0; r < 8; ++r) { m[r] = -1.0e30f; l[r] = 0.0f; }

  for (int kt = 0; kt < N_ / 32; ++kt) {
    // ---- issue ALL global loads for this key tile up front ----------------
    v16h bk[2][2];
    v16h bv[4];
    float kx[2], ky[2], kz[2];
    bool  vkey[2];
#pragma unroll
    for (int g = 0; g < 2; ++g) {
      int key = kt * 32 + g * 16 + l16;
      const _Float16* kr = Kbase + (size_t)key * DK_;
      bk[g][0] = *(const v16h*)(kr + hw * 16);
      bk[g][1] = *(const v16h*)(kr + 32 + hw * 16);
      const float* kc = cbase + (size_t)key * 3;
      kx[g] = kc[0]; ky[g] = kc[1]; kz[g] = kc[2];
      vkey[g] = (mbase[key] == 0);
    }
#pragma unroll
    for (int j = 0; j < 4; ++j)
      bv[j] = *(const v16h*)(Vbase + (size_t)(j * 16 + l16) * N_ + kt * 32 +
                             hw * 16);
    // prefetch the K/V stream two tiles ahead (global_prefetch_b8)
    if (kt + 2 < N_ / 32) {
      __builtin_prefetch(Kbase + (size_t)((kt + 2) * 32 + l16) * DK_, 0, 1);
      __builtin_prefetch(Vbase + (size_t)l16 * N_ + (kt + 2) * 32, 0, 1);
    }

    // ---- S = Q K^T (per 16-key group), + geometric bias & validity --------
    float s[2][8];
#pragma unroll
    for (int g = 0; g < 2; ++g) {
      v8f S = {};
      S = wmma16(aq0, bk[g][0], S);
      S = wmma16(aq1, bk[g][1], S);
#pragma unroll
      for (int r = 0; r < 8; ++r) {
        float dx = qcx[r] - kx[g], dy = qcy[r] - ky[g], dz = qcz[r] - kz[g];
        float d2 = dx * dx + dy * dy + dz * dz;
        bool valid = vkey[g] && (d2 >= d2min) && (d2 <= d2max);
        float lg = S[r] * 0.125f + betaC * d2;    // scale = 1/sqrt(64)
        s[g][r] = valid ? lg : -1.0e9f;
      }
    }

    // ---- online softmax (row = r + 8*hw; reductions within 16-lane halves)
    float alpha[8], mnew[8];
#pragma unroll
    for (int r = 0; r < 8; ++r) {
      float v = fmaxf(s[0][r], s[1][r]);
#pragma unroll
      for (int msk = 1; msk < 16; msk <<= 1) v = fmaxf(v, __shfl_xor(v, msk, 32));
      mnew[r]  = fmaxf(m[r], v);
      alpha[r] = __expf(m[r] - mnew[r]);
      m[r]     = mnew[r];
    }
#pragma unroll
    for (int r = 0; r < 8; ++r) {
      float p0 = __expf(s[0][r] - mnew[r]);
      float p1 = __expf(s[1][r] - mnew[r]);
      ldsP[wv][r + 8 * hw][l16]      = (_Float16)p0;
      ldsP[wv][r + 8 * hw][16 + l16] = (_Float16)p1;
      float rs = p0 + p1;
#pragma unroll
      for (int msk = 1; msk < 16; msk <<= 1) rs += __shfl_xor(rs, msk, 32);
      l[r] = l[r] * alpha[r] + rs;
#pragma unroll
      for (int j = 0; j < 4; ++j) O[j][r] *= alpha[r];
    }

    // ---- D-layout -> A-layout via per-wave LDS slice, then O += P V -------
    wave_lds_fence();
    v16h ap = load_a(&ldsP[wv][l16][0], hw);
#pragma unroll
    for (int j = 0; j < 4; ++j) O[j] = wmma16(ap, bv[j], O[j]);
  }

  // inter[b][n][dk*H + h]  (matches permute(0,2,3,1).reshape)
#pragma unroll
  for (int r = 0; r < 8; ++r) {
    float invl = 1.0f / l[r];
    int row = q0 + r + 8 * hw;
    _Float16* ob = inter + ((size_t)b * N_ + row) * D_ + h;
#pragma unroll
    for (int j = 0; j < 4; ++j)
      ob[(size_t)(j * 16 + l16) * H_] = (_Float16)(O[j][r] * invl);
  }
}

// ---------------------------------------------------------------------------
// Kernel 3: out = inter @ out_w^T   (4096x512 @ 512x512), fp32 result.
// ---------------------------------------------------------------------------
__global__ __launch_bounds__(128) void outproj_kernel(
    const _Float16* __restrict__ inter, const _Float16* __restrict__ w16,
    float* __restrict__ out) {
  const int lane = threadIdx.x & 31;
  const int wv   = threadIdx.x >> 5;
  const int hw   = lane >> 4;
  const int l16  = lane & 15;

  int id   = blockIdx.x * 4 + wv;          // 0 .. 2047
  int rt   = id & 255;                     // 256 row tiles
  int cb   = id >> 8;                      // 8 column blocks of 64
  int row0 = rt * 16;
  int col0 = cb * 64;

  v8f acc[4] = {};
  for (int c = 0; c < D_ / 32; ++c) {
    v16h bf[4];
#pragma unroll
    for (int j = 0; j < 4; ++j)
      bf[j] = *(const v16h*)(w16 + (size_t)(col0 + j * 16 + l16) * D_ +
                             c * 32 + hw * 16);
    v16h a = load_a(inter + ((size_t)row0 + l16) * D_ + c * 32, hw);
#pragma unroll
    for (int j = 0; j < 4; ++j) acc[j] = wmma16(a, bf[j], acc[j]);
  }
#pragma unroll
  for (int j = 0; j < 4; ++j)
#pragma unroll
    for (int r = 0; r < 8; ++r)
      out[((size_t)row0 + r + 8 * hw) * D_ + col0 + j * 16 + l16] = acc[j][r];
}

// ---------------------------------------------------------------------------
extern "C" void kernel_launch(void* const* d_in, const int* in_sizes, int n_in,
                              void* d_out, int out_size, void* d_ws,
                              size_t ws_size, hipStream_t stream) {
  const float* q      = (const float*)d_in[0];
  const float* k      = (const float*)d_in[1];
  const float* v      = (const float*)d_in[2];
  const float* coords = (const float*)d_in[3];
  const float* q_proj = (const float*)d_in[4];
  const float* k_proj = (const float*)d_in[5];
  const float* v_proj = (const float*)d_in[6];
  const float* q_bias = (const float*)d_in[7];
  const float* k_bias = (const float*)d_in[8];
  const float* v_bias = (const float*)d_in[9];
  const float* out_w  = (const float*)d_in[10];
  const float* sprd   = (const float*)d_in[11];
  const unsigned char* mask = (const unsigned char*)d_in[12];

  char* ws = (char*)d_ws;
  const size_t projT_b = (size_t)3 * H_ * DK_ * D_ * 2;   // 1.5 MB
  const size_t w16_b   = (size_t)D_ * D_ * 2;             // 0.5 MB
  const size_t qkv_b   = (size_t)B_ * H_ * N_ * DK_ * 2;  // 4 MB each

  _Float16* projT = (_Float16*)ws;
  _Float16* w16   = (_Float16*)(ws + projT_b);
  _Float16* Qh    = (_Float16*)(ws + projT_b + w16_b);
  _Float16* Kh    = (_Float16*)(ws + projT_b + w16_b + qkv_b);
  _Float16* VT    = (_Float16*)(ws + projT_b + w16_b + 2 * qkv_b);
  _Float16* inter = (_Float16*)(ws + projT_b + w16_b + 3 * qkv_b);

  convert_kernel<<<4096, 256, 0, stream>>>(q_proj, k_proj, v_proj, out_w,
                                           projT, w16);
  dim3 pg(512, 3);
  proj_kernel<<<pg, 128, 0, stream>>>(q, k, v, q_bias, k_bias, v_bias, projT,
                                      Qh, Kh, VT);
  attn_kernel<<<512, 128, 0, stream>>>(coords, sprd, mask, Qh, Kh, VT, inter);
  outproj_kernel<<<512, 128, 0, stream>>>(inter, w16, (float*)d_out);
}
